// RKN_65635690217734
// MI455X (gfx1250) — matrix operand order repaired and checked by reading, not compiled, not run on hardware
//
#include <hip/hip_runtime.h>

typedef __attribute__((ext_vector_type(16))) _Float16 v16h;
typedef __attribute__((ext_vector_type(8)))  float    v8f;

#define NB   128
#define NT   96
#define NM   (NB*NT)    // 12288 rows
#define NOBS 128
#define NENC 512
#define NLOD 64
#define NLSD 128
#define NK   16
#define NCH  128
#define NOUT 64
#define NE   2048       // banded basis columns: 4 blocks * 64 rows * 8 slots

__device__ __forceinline__ float elup1f(float x) { return x >= 0.f ? x + 1.f : __expf(x); }

// ---- WMMA fragment loaders (CDNA5 16x16x32 f16 layouts, wave32) ----
// A (16x32, MxK): lane holds row m=lane&15; lanes<16 use K-offsets kb+0..7 / kb+16..23
// with kb=0, lanes>=16 with kb=8.
__device__ __forceinline__ v16h load_a_f16(const _Float16* src, int ld, int row0, int k0, int lane) {
  int m  = row0 + (lane & 15);
  int kb = k0 + ((lane & 16) ? 8 : 0);
  const _Float16* p = src + (size_t)m * ld + kb;
  v16h a;
#pragma unroll
  for (int j = 0; j < 8; ++j) { a[j] = p[j]; a[j + 8] = p[16 + j]; }
  return a;
}
// B fragment from pre-packed (fragment-major) weights: one aligned 32B load per lane.
__device__ __forceinline__ v16h load_b_pk(const _Float16* pk, int ntiles, int kt, int nt, int lane) {
  const v16h* p = (const v16h*)(pk + (((size_t)kt * ntiles + nt) * 32 + lane) * 16);
  return *p;
}
#define WMMA_F32_F16(c, a, b) \
  __builtin_amdgcn_wmma_f32_16x16x32_f16(false, (a), false, (b), (short)0, (c), false, false)

// Pack one element of a KxN f32 weight matrix into WMMA-B fragment-major f16 layout.
__device__ __forceinline__ void pack_b(const float* W, _Float16* out, int N, int idx) {
  int ntiles = N >> 4;
  int tile = idx >> 9;              // 512 packed elems per tile
  int l    = (idx >> 4) & 31;
  int j    = idx & 15;
  int kt = tile / ntiles, nt = tile - kt * ntiles;
  int row = (kt << 5) + ((l & 16) ? 8 : 0) + (j & 7) + ((j & 8) ? 16 : 0);
  int col = (nt << 4) + (l & 15);
  out[idx] = (_Float16)W[row * N + col];
}

// ---- Kernel 0a: weight packing + banded basis + trans covar ----
__global__ void k_prep(const float* enc_W, const float* wmean_W, const float* wcov_W,
                       const float* cW1, const float* cW2, const float* dec_W,
                       const float* varh_W, const float* var_W,
                       const float* tm11, const float* tm12, const float* tm21, const float* tm22,
                       const float* log_tc,
                       _Float16* enc_Wp, _Float16* wmean_Wp, _Float16* wcov_Wp,
                       _Float16* cW1p, _Float16* cW2p, _Float16* dec_Wp,
                       _Float16* varh_Wp, _Float16* var_Wp, _Float16* basisMp, float* tc) {
  int idx = blockIdx.x * blockDim.x + threadIdx.x;
  if (idx < NOBS * NENC) pack_b(enc_W, enc_Wp, NENC, idx);
  if (idx < NENC * NLOD) { pack_b(wmean_W, wmean_Wp, NLOD, idx); pack_b(wcov_W, wcov_Wp, NLOD, idx); }
  if (idx < NLSD * NCH)  { pack_b(cW1, cW1p, NCH, idx); pack_b(varh_W, varh_Wp, NCH, idx); }
  if (idx < NCH * NK)    pack_b(cW2, cW2p, NK, idx);
  if (idx < NLSD * NOUT) pack_b(dec_W, dec_Wp, NOUT, idx);
  if (idx < NCH * NOUT)  pack_b(var_W, var_Wp, NOUT, idx);
  // Banded basis as WMMA-B pack: logical matrix is (K=32 zero-padded) x NE, single k-tile,
  // 128 n-tiles. K rows 0..15 = basis mixture index k; rows 16..31 = zero padding.
  if (idx < 32 * NE) {               // 65536 packed elements
    int nt = idx >> 9;               // tile = n-tile (kt==0)
    int l  = (idx >> 4) & 31;
    int j  = idx & 15;
    int krow = ((l & 16) ? 8 : 0) + (j & 7) + ((j & 8) ? 16 : 0);
    int col  = (nt << 4) + (l & 15); // e in [0, 2048)
    float v = 0.f;
    if (krow < NK) {
      int blk = col >> 9, r = (col >> 3) & 63, d = col & 7;
      int jj = r + d - 3;
      if (d < 7 && jj >= 0 && jj < 64) {
        const float* tm = (blk == 0) ? tm11 : (blk == 1) ? tm12 : (blk == 2) ? tm21 : tm22;
        v = tm[(krow * 64 + r) * 64 + jj];
      }
    }
    basisMp[idx] = (_Float16)v;
  }
  if (idx < NLSD) tc[idx] = elup1f(log_tc[idx]);
}

// ---- Kernel 0b: obs f32 -> f16 (single conversion instead of per-tile) ----
__global__ void k_cvt_obs(const float* obs, _Float16* obs_h) {
  int idx = blockIdx.x * blockDim.x + threadIdx.x;
  obs_h[idx] = (_Float16)obs[idx];
}

// ---- Kernel 1: h = relu(obs @ enc_W + enc_b); each wave does 4 N-tiles, A reused ----
__global__ void __launch_bounds__(256) k_encode(const _Float16* obs_h, const _Float16* enc_Wp,
                                                const float* enc_b, _Float16* h) {
  int lane = threadIdx.x & 31, wv = threadIdx.x >> 5;
  int m0 = blockIdx.x * 16;
  v8f c[4] = {};
  for (int kt = 0; kt < NOBS / 32; ++kt) {
    v16h a = load_a_f16(obs_h, NOBS, m0, kt * 32, lane);
#pragma unroll
    for (int q = 0; q < 4; ++q) {
      v16h b = load_b_pk(enc_Wp, NENC / 16, kt, wv * 4 + q, lane);
      c[q] = WMMA_F32_F16(c[q], a, b);
    }
  }
  int mb = m0 + ((lane & 16) ? 8 : 0);
#pragma unroll
  for (int q = 0; q < 4; ++q) {
    int n = (wv * 4 + q) * 16 + (lane & 15);
    float bias = enc_b[n];
#pragma unroll
    for (int r = 0; r < 8; ++r) {
      float v = c[q][r] + bias;
      h[(size_t)(mb + r) * NENC + n] = (_Float16)fmaxf(v, 0.f);
    }
  }
}

// ---- Kernel 2: wm = normalize(h@wmean_W + b), wc = elup1(h@wcov_W + b); time-major ----
// 4 waves; each wave computes the wm AND wc tile for its N range (shared A fragment).
__global__ void __launch_bounds__(128) k_wmwc(const _Float16* h, const _Float16* wmean_Wp,
                                              const float* wmean_b, const _Float16* wcov_Wp,
                                              const float* wcov_b, float* wm_g, float* wc_g) {
  __shared__ float s_wm[16][64];
  __shared__ float s_wc[16][64];
  __shared__ float s_rn[16];
  int lane = threadIdx.x & 31, wv = threadIdx.x >> 5;  // wv in 0..3
  int m0 = blockIdx.x * 16;
  v8f cm = {}, cc = {};
  for (int kt = 0; kt < NENC / 32; ++kt) {
    v16h a = load_a_f16(h, NENC, m0, kt * 32, lane);
    v16h b1 = load_b_pk(wmean_Wp, NLOD / 16, kt, wv, lane);
    cm = WMMA_F32_F16(cm, a, b1);
    v16h b2 = load_b_pk(wcov_Wp, NLOD / 16, kt, wv, lane);
    cc = WMMA_F32_F16(cc, a, b2);
  }
  int n  = wv * 16 + (lane & 15);
  int rb = (lane & 16) ? 8 : 0;
  float bm = wmean_b[n], bc = wcov_b[n];
#pragma unroll
  for (int r = 0; r < 8; ++r) {
    s_wm[rb + r][n] = cm[r] + bm;
    s_wc[rb + r][n] = cc[r] + bc;
  }
  __syncthreads();
  if (threadIdx.x < 16) {
    float s = 0.f;
    for (int j = 0; j < 64; ++j) { float v = s_wm[threadIdx.x][j]; s += v * v; }
    s_rn[threadIdx.x] = rsqrtf(s);
  }
  __syncthreads();
  for (int p = threadIdx.x; p < 16 * 64; p += 128) {
    int r = p >> 6, col = p & 63;
    int m = m0 + r;
    int b = m / NT, t = m % NT;
    size_t o = (size_t)t * NB * NLOD + (size_t)b * NLOD + col;
    wm_g[o] = s_wm[r][col] * s_rn[r];
    wc_g[o] = elup1f(s_wc[r][col]);
  }
}

// ---- Kernel 3: sequential scan; 8 blocks x 16 batch rows, state in LDS ----
__global__ void __launch_bounds__(256) k_scan(const _Float16* cW1p, const float* cb1,
                                              const _Float16* cW2p, const float* cb2,
                                              const _Float16* basisMp, const float* tc,
                                              const float* wm_g, const float* wc_g,
                                              _Float16* post_h) {
  __shared__ float    s_mean[16][NLSD];
  __shared__ _Float16 s_mean_h[16][NLSD];   // f16 mirror for Phase-1 A fragments
  __shared__ float    s_cu[16][NLOD];
  __shared__ float    s_cl[16][NLOD];
  __shared__ float    s_cs[16][NLOD];
  __shared__ _Float16 s_hh[16][NCH];
  __shared__ float    s_logit[16][16];
  __shared__ float    s_coeff[16][16];
  __shared__ _Float16 s_Tm[16][NE];

  int lane = threadIdx.x & 31, wv = threadIdx.x >> 5;
  int bb = blockIdx.x * 16;

  for (int p = threadIdx.x; p < 16 * NLSD; p += 256) {
    s_mean[p >> 7][p & 127] = 0.f;
    s_mean_h[p >> 7][p & 127] = (_Float16)0.f;
  }
  for (int p = threadIdx.x; p < 16 * NLOD; p += 256) {
    s_cu[p >> 6][p & 63] = 10.f; s_cl[p >> 6][p & 63] = 10.f; s_cs[p >> 6][p & 63] = 0.f;
  }
  __syncthreads();

  for (int t = 0; t < NT; ++t) {
    // Prefetch this step's wm/wc rows (latency-sensitive Phase 4 consumer).
    {
      size_t wo = (size_t)t * NB * NLOD + (size_t)bb * NLOD + (size_t)threadIdx.x * 4;
      __builtin_prefetch(wm_g + wo, 0, 0);
      __builtin_prefetch(wc_g + wo, 0, 0);
    }
    // Phase 1: hh = relu(mean @ cW1 + cb1)  (16x128), one N-tile per wave
    {
      v8f c = {};
      for (int kt = 0; kt < NLSD / 32; ++kt) {
        v16h a = load_a_f16(&s_mean_h[0][0], NLSD, 0, kt * 32, lane);
        v16h b = load_b_pk(cW1p, NCH / 16, kt, wv, lane);
        c = WMMA_F32_F16(c, a, b);
      }
      int n  = wv * 16 + (lane & 15);
      int rb = (lane & 16) ? 8 : 0;
      float bs = cb1[n];
#pragma unroll
      for (int r = 0; r < 8; ++r) s_hh[rb + r][n] = (_Float16)fmaxf(c[r] + bs, 0.f);
    }
    __syncthreads();
    // Phase 2: logits = hh @ cW2 + cb2  (16x16), wave 0 only
    if (wv == 0) {
      v8f c = {};
      for (int kt = 0; kt < NCH / 32; ++kt) {
        v16h a = load_a_f16(&s_hh[0][0], NCH, 0, kt * 32, lane);
        v16h b = load_b_pk(cW2p, 1, kt, 0, lane);
        c = WMMA_F32_F16(c, a, b);
      }
      int n  = lane & 15;
      int rb = (lane & 16) ? 8 : 0;
      float bs = cb2[n];
#pragma unroll
      for (int r = 0; r < 8; ++r) s_logit[rb + r][n] = c[r] + bs;
    }
    __syncthreads();
    if (threadIdx.x < 16) {
      int r = threadIdx.x;
      float mx = s_logit[r][0];
      for (int j = 1; j < 16; ++j) mx = fmaxf(mx, s_logit[r][j]);
      float sum = 0.f, e[16];
      for (int j = 0; j < 16; ++j) { e[j] = __expf(s_logit[r][j] - mx); sum += e[j]; }
      float inv = 1.f / sum;
      for (int j = 0; j < 16; ++j) s_coeff[r][j] = e[j] * inv;
    }
    __syncthreads();
    // Phase 3: Tm = coeff @ basisM  (16 x 2048), K=16 zero-padded to 32 (pad baked into pack)
    {
      v16h a;
      {
        int m  = lane & 15;
        int kb = (lane & 16) ? 8 : 0;
#pragma unroll
        for (int j = 0; j < 8; ++j) { a[j] = (_Float16)s_coeff[m][kb + j]; a[j + 8] = (_Float16)0.f; }
      }
      for (int it = 0; it < 16; ++it) {
        int nt = wv * 16 + it;
        v16h b = load_b_pk(basisMp, 128, 0, nt, lane);
        v8f c = {};
        c = WMMA_F32_F16(c, a, b);
        int n  = nt * 16 + (lane & 15);
        int rb = (lane & 16) ? 8 : 0;
#pragma unroll
        for (int r = 0; r < 8; ++r) s_Tm[rb + r][n] = (_Float16)c[r];
      }
    }
    __syncthreads();
    // Phase 4: banded mean/covariance propagation + Kalman update
    float r_pmu[4], r_pml[4], r_pcu[4], r_pcl[4], r_pcs[4];
#pragma unroll
    for (int p = 0; p < 4; ++p) {
      int pair = threadIdx.x + p * 256;   // 0..1023 -> (b in 0..15, i in 0..63)
      int b = pair >> 6, i = pair & 63;
      float nmu = 0.f, nml = 0.f, ncu = 0.f, ncl = 0.f, ncs = 0.f;
#pragma unroll
      for (int d = 0; d < 7; ++d) {
        int j = i + d - 3;
        if (j < 0 || j >= 64) continue;
        int e = i * 8 + d;
        float a11 = (float)s_Tm[b][e];
        float a12 = (float)s_Tm[b][512 + e];
        float a21 = (float)s_Tm[b][1024 + e];
        float a22 = (float)s_Tm[b][1536 + e];
        float mu = s_mean[b][j], ml = s_mean[b][64 + j];
        float cu = s_cu[b][j], cl = s_cl[b][j], cs = s_cs[b][j];
        nmu += a11 * mu + a12 * ml;
        nml += a21 * mu + a22 * ml;
        ncu += a11 * a11 * cu + 2.f * a11 * a12 * cs + a12 * a12 * cl;
        ncl += a21 * a21 * cu + 2.f * a21 * a22 * cs + a22 * a22 * cl;
        ncs += a21 * a11 * cu + a22 * a11 * cs + a21 * a12 * cs + a22 * a12 * cl;
      }
      ncu += tc[i];
      ncl += tc[64 + i];
      size_t wo = (size_t)t * NB * NLOD + (size_t)(bb + b) * NLOD + i;
      float wc_t = wc_g[wo], wm_t = wm_g[wo];
      float denom = ncu + wc_t;
      float qu = ncu / denom, ql = ncs / denom;
      float res = wm_t - nmu;
      float pmu = nmu + qu * res, pml = nml + ql * res;
      float cf = 1.f - qu;
      r_pmu[p] = pmu; r_pml[p] = pml;
      r_pcu[p] = cf * ncu; r_pcl[p] = ncl - ql * ncs; r_pcs[p] = cf * ncs;
      size_t po = ((size_t)(bb + b) * NT + t) * NLSD;
      post_h[po + i]      = (_Float16)pmu;
      post_h[po + 64 + i] = (_Float16)pml;
    }
    __syncthreads();
#pragma unroll
    for (int p = 0; p < 4; ++p) {
      int pair = threadIdx.x + p * 256;
      int b = pair >> 6, i = pair & 63;
      s_mean[b][i] = r_pmu[p];            s_mean[b][64 + i] = r_pml[p];
      s_mean_h[b][i] = (_Float16)r_pmu[p]; s_mean_h[b][64 + i] = (_Float16)r_pml[p];
      s_cu[b][i] = r_pcu[p];   s_cl[b][i] = r_pcl[p];   s_cs[b][i] = r_pcs[p];
    }
    __syncthreads();
  }
}

// ---- Kernel 4: decoder; 4 waves, shared-A multi-tile ----
__global__ void __launch_bounds__(128) k_decode(const _Float16* post_h,
                                                const _Float16* dec_Wp, const float* dec_b,
                                                const _Float16* varh_Wp, const float* varh_b,
                                                const _Float16* var_Wp, const float* var_b,
                                                float* out) {
  __shared__ _Float16 s_vh[16][NCH];
  int lane = threadIdx.x & 31, wv = threadIdx.x >> 5;  // wv in 0..3
  int m0 = blockIdx.x * 16;
  int rb = (lane & 16) ? 8 : 0;
  int mb = m0 + rb;
  // Phase A: vh = relu(post @ varh_W + varh_b); each wave 2 N-tiles, plus pred_mean tile
  {
    v8f c0 = {}, c1 = {}, cpm = {};
    for (int kt = 0; kt < NLSD / 32; ++kt) {
      v16h a = load_a_f16(post_h, NLSD, m0, kt * 32, lane);
      v16h b0 = load_b_pk(varh_Wp, NCH / 16, kt, wv * 2, lane);
      c0 = WMMA_F32_F16(c0, a, b0);
      v16h b1 = load_b_pk(varh_Wp, NCH / 16, kt, wv * 2 + 1, lane);
      c1 = WMMA_F32_F16(c1, a, b1);
      v16h b2 = load_b_pk(dec_Wp, NOUT / 16, kt, wv, lane);
      cpm = WMMA_F32_F16(cpm, a, b2);
    }
    int n0 = wv * 32 + (lane & 15);
    int n1 = n0 + 16;
    float b0 = varh_b[n0], b1 = varh_b[n1];
#pragma unroll
    for (int r = 0; r < 8; ++r) {
      s_vh[rb + r][n0] = (_Float16)fmaxf(c0[r] + b0, 0.f);
      s_vh[rb + r][n1] = (_Float16)fmaxf(c1[r] + b1, 0.f);
    }
    int nm = wv * 16 + (lane & 15);
    float bm = dec_b[nm];
#pragma unroll
    for (int r = 0; r < 8; ++r) out[(size_t)(mb + r) * 128 + nm] = cpm[r] + bm;
  }
  __syncthreads();
  // Phase B: pred_var = elup1(vh @ var_W + var_b); one tile per wave
  {
    v8f c = {};
    for (int kt = 0; kt < NCH / 32; ++kt) {
      v16h a = load_a_f16(&s_vh[0][0], NCH, 0, kt * 32, lane);
      v16h b = load_b_pk(var_Wp, NOUT / 16, kt, wv, lane);
      c = WMMA_F32_F16(c, a, b);
    }
    int n = wv * 16 + (lane & 15);
    float bs = var_b[n];
#pragma unroll
    for (int r = 0; r < 8; ++r) out[(size_t)(mb + r) * 128 + 64 + n] = elup1f(c[r] + bs);
  }
}

extern "C" void kernel_launch(void* const* d_in, const int* in_sizes, int n_in,
                              void* d_out, int out_size, void* d_ws, size_t ws_size,
                              hipStream_t stream) {
  const float* obs      = (const float*)d_in[0];
  const float* enc_W    = (const float*)d_in[1];
  const float* enc_b    = (const float*)d_in[2];
  const float* wmean_W  = (const float*)d_in[3];
  const float* wmean_b  = (const float*)d_in[4];
  const float* wcov_W   = (const float*)d_in[5];
  const float* wcov_b   = (const float*)d_in[6];
  const float* cW1      = (const float*)d_in[7];
  const float* cb1      = (const float*)d_in[8];
  const float* cW2      = (const float*)d_in[9];
  const float* cb2      = (const float*)d_in[10];
  const float* tm11     = (const float*)d_in[11];
  const float* tm12     = (const float*)d_in[12];
  const float* tm21     = (const float*)d_in[13];
  const float* tm22     = (const float*)d_in[14];
  const float* log_tc   = (const float*)d_in[15];
  const float* dec_W    = (const float*)d_in[16];
  const float* dec_b    = (const float*)d_in[17];
  const float* varh_W   = (const float*)d_in[18];
  const float* varh_b   = (const float*)d_in[19];
  const float* var_W    = (const float*)d_in[20];
  const float* var_b    = (const float*)d_in[21];
  float* out = (float*)d_out;

  char* w = (char*)d_ws;
  size_t off = 0;
  auto alloc = [&](size_t bytes) -> void* {
    void* p = (void*)(w + off);
    off = (off + bytes + 255) & ~(size_t)255;
    return p;
  };
  _Float16* h_h      = (_Float16*)alloc((size_t)NM * NENC * 2);
  _Float16* obs_h    = (_Float16*)alloc((size_t)NM * NOBS * 2);
  _Float16* enc_Wp   = (_Float16*)alloc((size_t)NOBS * NENC * 2);
  _Float16* wmean_Wp = (_Float16*)alloc((size_t)NENC * NLOD * 2);
  _Float16* wcov_Wp  = (_Float16*)alloc((size_t)NENC * NLOD * 2);
  _Float16* cW1p     = (_Float16*)alloc((size_t)NLSD * NCH * 2);
  _Float16* cW2p     = (_Float16*)alloc((size_t)NCH * NK * 2);
  _Float16* dec_Wp   = (_Float16*)alloc((size_t)NLSD * NOUT * 2);
  _Float16* varh_Wp  = (_Float16*)alloc((size_t)NLSD * NCH * 2);
  _Float16* var_Wp   = (_Float16*)alloc((size_t)NCH * NOUT * 2);
  _Float16* basisMp  = (_Float16*)alloc((size_t)32 * NE * 2);   // K padded to 32
  float*    tc       = (float*)alloc((size_t)NLSD * 4);
  float*    wm_g     = (float*)alloc((size_t)NT * NB * NLOD * 4);
  float*    wc_g     = (float*)alloc((size_t)NT * NB * NLOD * 4);
  _Float16* post_h   = (_Float16*)alloc((size_t)NM * NLSD * 2);

  k_prep<<<256, 256, 0, stream>>>(enc_W, wmean_W, wcov_W, cW1, cW2, dec_W, varh_W, var_W,
                                  tm11, tm12, tm21, tm22, log_tc,
                                  enc_Wp, wmean_Wp, wcov_Wp, cW1p, cW2p, dec_Wp,
                                  varh_Wp, var_Wp, basisMp, tc);
  k_cvt_obs<<<(NM * NOBS) / 256, 256, 0, stream>>>(obs, obs_h);
  k_encode<<<NM / 16, 256, 0, stream>>>(obs_h, enc_Wp, enc_b, h_h);
  k_wmwc<<<NM / 16, 128, 0, stream>>>(h_h, wmean_Wp, wmean_b, wcov_Wp, wcov_b, wm_g, wc_g);
  k_scan<<<NB / 16, 256, 0, stream>>>(cW1p, cb1, cW2p, cb2, basisMp, tc, wm_g, wc_g, post_h);
  k_decode<<<NM / 16, 128, 0, stream>>>(post_h, dec_Wp, dec_b, varh_Wp, varh_b, var_Wp, var_b, out);
}
